// feature_extraction_48558900248649
// MI455X (gfx1250) — matrix operand order repaired and checked by reading, not compiled, not run on hardware
//
#include <hip/hip_runtime.h>
#include <hip/hip_bf16.h>

// ---------------------------------------------------------------------------
// Problem constants (from the reference)
// ---------------------------------------------------------------------------
#define B_     64
#define C_     6
#define T_     32768
#define CHUNK_ 256
#define FL_    8
#define E_     256
#define NCH_   128              // T/CHUNK
#define L_     (NCH_ * FL_)     // 1024
#define KDIM   (C_ * CHUNK_)    // 1536
#define NDIM   (E_ * FL_)       // 2048
#define KT     (KDIM / 32)      // 48 K-steps of 32 (bf16 WMMA K), even

#define LDSTRIDE 20             // dwords per 32-bf16 row (16B-aligned, bank-spread)

typedef __attribute__((ext_vector_type(16))) __bf16 v16bf;
typedef __attribute__((ext_vector_type(2)))  __bf16 v2bf;
typedef __attribute__((ext_vector_type(8)))  float  v8f;

struct __attribute__((aligned(16))) U4 { unsigned a, b, c, d; };
union Frag { U4 q[2]; v16bf v; };

// f32 pair -> packed bf16 dword.  Native converter if declared; otherwise
// round-half-up + hi-half merge (v_add x2 + v_perm_b32 pattern).
__device__ __forceinline__ unsigned pk_bf16(float a, float b) {
#if __has_builtin(__builtin_amdgcn_cvt_pk_bf16_f32)
  union { v2bf v; unsigned u; } c;
  c.v = __builtin_amdgcn_cvt_pk_bf16_f32(a, b);
  return c.u;
#else
  const unsigned ra = __float_as_uint(a) + 0x8000u;
  const unsigned rb = __float_as_uint(b) + 0x8000u;
  return (ra >> 16) | (rb & 0xFFFF0000u);   // byte-select: v_perm_b32
#endif
}

// ---------------------------------------------------------------------------
// Kernel 1: chunk validity.  One wave32 per chunk (8 chunks / 256-thread block)
// valid(b,n) = any(input_mask[b, n*256 : (n+1)*256] != 0)
// Writes: rowscale[b*128+n] (float 0/1, in d_ws) and output_mask[b, n*8+f].
// ---------------------------------------------------------------------------
__global__ __launch_bounds__(256) void mask_kernel(const int* __restrict__ im,
                                                   float* __restrict__ rowscale,
                                                   float* __restrict__ out_mask) {
  const int lane  = threadIdx.x & 31;
  const int wave  = threadIdx.x >> 5;
  const int chunk = blockIdx.x * 8 + wave;      // 0 .. B*NCH-1
  const int b = chunk >> 7, n = chunk & 127;
  const int* p = im + ((size_t)b << 15) + n * CHUNK_;
  int any = 0;
#pragma unroll
  for (int i = 0; i < 8; ++i) any |= p[lane + i * 32];
  const float s = (__ballot(any != 0) != 0ull) ? 1.0f : 0.0f;
  if (lane == 0) rowscale[chunk] = s;
  if (lane < FL_) out_mask[b * L_ + n * FL_ + lane] = s;
}

// ---------------------------------------------------------------------------
// Kernel 2: fused GEMM + bias + mask + (E,FL) de-interleave.
//   grid = (NDIM/128, B), block = 256 (8 wave32).
//   Tile: 128 rows (chunks of batch b) x 128 permuted columns, K-steps of 32.
//   Wave (wm = wave/4 in {0,1}, wn = wave%4) owns a 64x32 sub-tile:
//   4 A-frags x 2 B-frags -> 8 accumulators of v_wmma_f32_16x16x32_bf16.
//   Column permutation col' = f*16 + e_local makes the final stores coalesced.
// ---------------------------------------------------------------------------
__global__ __launch_bounds__(256) void extractor_gemm(
    const float* __restrict__ x, const float* __restrict__ W,
    const float* __restrict__ bias, const float* __restrict__ rowscale,
    float* __restrict__ out) {
  __shared__ __align__(16) unsigned sA[2][128 * LDSTRIDE];
  __shared__ __align__(16) unsigned sB[2][128 * LDSTRIDE + 32];

  const int tid   = threadIdx.x;
  const int lane  = tid & 31;
  const int wave  = tid >> 5;
  const int wm    = wave >> 2;        // 0..1 : 64-row half of tile
  const int wn    = wave & 3;         // 0..3 : 32-col' slice of tile
  const int m16   = lane & 15;
  const int half  = lane >> 4;
  const int b     = blockIdx.y;       // one batch per M-tile (M = b*128 + n)
  const int tileN = blockIdx.x * 128; // d-range of this tile

  v8f acc[4][2];
#pragma unroll
  for (int i = 0; i < 4; ++i)
#pragma unroll
    for (int j = 0; j < 2; ++j) acc[i][j] = (v8f)0.0f;

  // ---- stage one 128x32 A tile (f32->bf16) and 32x128 B tile into LDS ----
  // buf is always a compile-time constant at the call sites below.
  auto load_tile = [&](int kt, int buf) {
    const int k0   = kt * 32;
    const int c    = k0 >> 8;          // which channel this K-slice lives in
    const int toff = k0 & 255;
    const float* xb = x + ((size_t)(b * C_ + c) << 15) + toff;
#pragma unroll
    for (int it = 0; it < 4; ++it) {          // 128 rows x 8 float4 groups
      const int idx = tid + it * 256;
      const int r = idx >> 3, g = idx & 7;    // coalesced 16B per lane
      const float4 v = *reinterpret_cast<const float4*>(xb + r * CHUNK_ + g * 4);
      unsigned* dst = &sA[buf][r * LDSTRIDE + g * 2];
      const unsigned p0 = pk_bf16(v.x, v.y);
      const unsigned p1 = pk_bf16(v.z, v.w);
      dst[0] = p0; dst[1] = p1;               // 8B store, 8B aligned
    }
    const float* wb = W + (size_t)k0 * NDIM + tileN;
#pragma unroll
    for (int it = 0; it < 8; ++it) {          // 16 packed-k rows x 128 cols
      const int idx = tid + it * 256;
      const int dl = idx & 127, kp = idx >> 7;        // dl coalesced in global
      const float w0 = wb[(2 * kp) * NDIM + dl];
      const float w1 = wb[(2 * kp + 1) * NDIM + dl];
      const int colp = ((dl & 7) << 4) + (dl >> 3);   // col' = f*16 + e_local
      sB[buf][colp * LDSTRIDE + ((colp >> 4) << 2) + kp] = pk_bf16(w0, w1);
    }
  };

  // ---- 8x v_wmma_f32_16x16x32_bf16 per K-step ----
  auto compute_tile = [&](int buf) {
    const unsigned* pa = sA[buf];
    const unsigned* pb = sB[buf];
    v16bf afr[4], bfr[2];
#pragma unroll
    for (int i = 0; i < 4; ++i) {
      // A 16x32 bf16 layout: lane m=lane%16; half0 K=0..7,16..23 ; half1 +8
      const int row = wm * 64 + i * 16 + m16;
      const unsigned* p = pa + row * LDSTRIDE + half * 4;
      Frag u;
      u.q[0] = *reinterpret_cast<const U4*>(p);       // ds_load_b128
      u.q[1] = *reinterpret_cast<const U4*>(p + 8);
      afr[i] = u.v;
    }
#pragma unroll
    for (int j = 0; j < 2; ++j) {
      // B 32x16 bf16 layout: lane n=lane%16; half0 K=0..7,16..23 ; half1 +8
      const int colp = wn * 32 + j * 16 + m16;
      const unsigned* p = pb + colp * LDSTRIDE + ((colp >> 4) << 2) + half * 4;
      Frag u;
      u.q[0] = *reinterpret_cast<const U4*>(p);
      u.q[1] = *reinterpret_cast<const U4*>(p + 8);
      bfr[j] = u.v;
    }
#pragma unroll
    for (int i = 0; i < 4; ++i)
#pragma unroll
      for (int j = 0; j < 2; ++j)
        acc[i][j] = __builtin_amdgcn_wmma_f32_16x16x32_bf16(
            false, afr[i], false, bfr[j], (short)0, acc[i][j], false, false);
  };

  // ---- software-pipelined K loop, unrolled x2 so LDS buffer indices are
  //      compile-time constants (offsets fold into instruction immediates) ----
  load_tile(0, 0);
  __syncthreads();
#pragma unroll 1
  for (int kt = 0; kt < KT; kt += 2) {
    load_tile(kt + 1, 1);           // KT even and kt+1 <= KT-1: always valid
    compute_tile(0);
    __syncthreads();
    if (kt + 2 < KT) load_tile(kt + 2, 0);
    compute_tile(1);
    __syncthreads();
  }

  // ---- epilogue: bias + mask + permuted (coalesced) store ----
  // col' = wn*32 + j*16 + m16  ->  f = wn*2 + j (const per vgpr), e_local = m16
  // out[b, n*8+f, e] with e = blockIdx.x*16 + m16 : 16 lanes -> 64B contiguous
  const int e = blockIdx.x * 16 + m16;
  float* outb = out + ((size_t)b << 18);     // b * L_ * E_
#pragma unroll
  for (int j = 0; j < 2; ++j) {
    const int f = wn * 2 + j;
    const float bv = bias[tileN + m16 * 8 + f];
#pragma unroll
    for (int i = 0; i < 4; ++i) {
#pragma unroll
      for (int v = 0; v < 8; ++v) {
        const int n = wm * 64 + i * 16 + half * 8 + v;  // C/D layout row map
        const float s = rowscale[b * NCH_ + n];
        outb[(size_t)(n * FL_ + f) * E_ + e] = (acc[i][j][v] + bv) * s;
      }
    }
  }
}

// ---------------------------------------------------------------------------
extern "C" void kernel_launch(void* const* d_in, const int* in_sizes, int n_in,
                              void* d_out, int out_size, void* d_ws, size_t ws_size,
                              hipStream_t stream) {
  const float* x          = (const float*)d_in[0];
  const int*   input_mask = (const int*)d_in[1];
  const float* W          = (const float*)d_in[2];
  const float* bias       = (const float*)d_in[3];
  float* out      = (float*)d_out;
  float* out_mask = out + (size_t)B_ * L_ * E_;   // second tuple element
  float* rowscale = (float*)d_ws;                 // B*NCH floats of scratch

  mask_kernel<<<dim3((B_ * NCH_) / 8), dim3(256), 0, stream>>>(
      input_mask, rowscale, out_mask);
  extractor_gemm<<<dim3(NDIM / 128, B_), dim3(256), 0, stream>>>(
      x, W, bias, rowscale, out);
}
